// DynamicCRF_55336358641969
// MI455X (gfx1250) — matrix operand order repaired and checked by reading, not compiled, not run on hardware
//
#include <hip/hip_runtime.h>
#include <math.h>

#define BB   16
#define TT   256
#define VV   32000
#define RNK  32
#define BM   64
#define CAP  8192   // candidate capacity (expected ~730 for N(0,1); 8192 is >100 sigma)

typedef __attribute__((ext_vector_type(16))) __bf16 v16bf;
typedef __attribute__((ext_vector_type(8)))  float  v8f;

__device__ __forceinline__ unsigned key_of(float f) {
    unsigned u = __float_as_uint(f);
    return (u & 0x80000000u) ? ~u : (u | 0x80000000u);  // monotone: larger float -> larger key
}

__device__ __forceinline__ float key_inv(unsigned k) {
    unsigned u = (k & 0x80000000u) ? (k ^ 0x80000000u) : ~k;
    return __uint_as_float(u);
}

__device__ __forceinline__ unsigned short f2bf(float x) {  // RNE float -> bf16 bits
    unsigned u = __float_as_uint(x);
    unsigned r = u + 0x7FFFu + ((u >> 16) & 1u);
    return (unsigned short)(r >> 16);
}

// ---------------------------------------------------------------------------
// Kernel 1: exact top-64 per (b,t) row.
//  Sweep A: per-wave top-byte histograms (privatized -> conflict chains <= 32).
//  Sweep B: keys with top byte > pivot go straight to the beam; keys with top
//           byte == pivot become candidates in LDS (~730 expected).
//  Then 3 radix passes + collection over candidates only.
// Beam ORDER is irrelevant (logsumexp over the beam axis is permutation
// invariant); only the set matters. Target forced in via key = 0xFFFFFFFF.
// ---------------------------------------------------------------------------
__global__ __launch_bounds__(256) void topk64_kernel(
    const float* __restrict__ em, const int* __restrict__ targets,
    int* __restrict__ beam_tgts, float* __restrict__ beam_em)
{
    __shared__ unsigned whist[8][256];
    __shared__ unsigned hist[256];
    __shared__ unsigned candKey[CAP];
    __shared__ int      candIdx[CAP];
    __shared__ unsigned sPrefVal, sPrefMask, sRemain, sCntG, sCntE, sCand;
    __shared__ float    sTgtVal;

    const int row  = blockIdx.x;                 // row = b*T + t
    const int tid  = threadIdx.x;
    const int wave = tid >> 5;
    const float4* __restrict__ rp4 = (const float4*)(em + (size_t)row * VV);
    const int tgt = targets[row];

    if (tid == 0) { sCntG = 0u; sCntE = 0u; sCand = 0u; sTgtVal = 0.f; }
    for (int i = tid; i < 8 * 256; i += 256) ((unsigned*)whist)[i] = 0u;
    __syncthreads();

    // ---- Sweep A: histogram of key top byte (per-wave privatized) ----
    for (int i = tid; i < VV / 4; i += 256) {
        if (i + 1024 < VV / 4) __builtin_prefetch(rp4 + i + 1024, 0, 1);
        float4 v = rp4[i];
        const float fv[4] = {v.x, v.y, v.z, v.w};
#pragma unroll
        for (int j = 0; j < 4; ++j) {
            const int idx = i * 4 + j;
            unsigned k = (idx == tgt) ? 0xFFFFFFFFu : key_of(fv[j]);
            atomicAdd(&whist[wave][k >> 24], 1u);
        }
    }
    __syncthreads();
    {   // combine 8 wave histograms (one bin per thread)
        unsigned tot = 0u;
#pragma unroll
        for (int w = 0; w < 8; ++w) tot += whist[w][tid];
        hist[tid] = tot;
    }
    __syncthreads();
    if (tid == 0) {
        unsigned cum = 0u; int bsel = 0;
        for (int b = 255; b >= 0; --b) {
            unsigned h = hist[b];
            if (cum + h >= (unsigned)BM) { bsel = b; break; }
            cum += h;
        }
        sPrefVal  = (unsigned)bsel << 24;
        sPrefMask = 0xFF000000u;
        sRemain   = (unsigned)BM - cum;    // still needed among pivot-byte keys
    }
    __syncthreads();
    const unsigned selByte = sPrefVal >> 24;

    // ---- Sweep B: emit sure winners, collect pivot-byte candidates ----
    for (int i = tid; i < VV / 4; i += 256) {
        float4 v = rp4[i];
        const float fv[4] = {v.x, v.y, v.z, v.w};
#pragma unroll
        for (int j = 0; j < 4; ++j) {
            const int idx = i * 4 + j;
            const float f = fv[j];
            unsigned k = key_of(f);
            if (idx == tgt) { k = 0xFFFFFFFFu; sTgtVal = f; }
            const unsigned tb = k >> 24;
            if (tb > selByte) {
                int pos = (int)atomicAdd(&sCntG, 1u);
                beam_tgts[row * BM + pos] = idx;
                beam_em  [row * BM + pos] = f;
            } else if (tb == selByte) {
                unsigned c = atomicAdd(&sCand, 1u);
                if (c < CAP) { candKey[c] = k; candIdx[c] = idx; }
            }
        }
    }
    __syncthreads();
    const int nCand = (int)(sCand < (unsigned)CAP ? sCand : (unsigned)CAP);

    // ---- 3 refining radix passes over candidates (bytes 2..0) ----
    for (int p = 2; p >= 0; --p) {
        hist[tid] = 0u;
        __syncthreads();
        const unsigned pv = sPrefVal, pm = sPrefMask;
        for (int c = tid; c < nCand; c += 256) {
            unsigned k = candKey[c];
            if ((k & pm) == pv) atomicAdd(&hist[(k >> (8 * p)) & 255u], 1u);
        }
        __syncthreads();
        if (tid == 0) {
            unsigned rem = sRemain, cum = 0u; int bsel = 0;
            for (int b = 255; b >= 0; --b) {
                unsigned h = hist[b];
                if (cum + h >= rem) { bsel = b; break; }
                cum += h;
            }
            sRemain   = rem - cum;
            sPrefVal  = pv | ((unsigned)bsel << (8 * p));
            sPrefMask = pm | (0xFFu << (8 * p));
        }
        __syncthreads();
    }
    const unsigned Kth = sPrefVal;    // exact key of the 64th-largest element
    const unsigned rem = sRemain;     // ties at Kth still to take

    // ---- collection over candidates ----
    for (int c = tid; c < nCand; c += 256) {
        const unsigned k = candKey[c];
        int pos = -1;
        if (k > Kth) {
            pos = (int)atomicAdd(&sCntG, 1u);     // continues after sweep-B winners
        } else if (k == Kth) {
            unsigned e = atomicAdd(&sCntE, 1u);
            if (e < rem) pos = (int)(BM - rem + e);
        }
        if (pos >= 0) {
            beam_tgts[row * BM + pos] = candIdx[c];
            beam_em  [row * BM + pos] = (k == 0xFFFFFFFFu) ? sTgtVal : key_inv(k);
        }
    }
}

// ---------------------------------------------------------------------------
// Kernel 2: numerator[b] = sum_t mask * (em[b,t,tgt] + (t>0)*dot32(E1[tgt_{t-1}],E2[tgt_t]))
// ---------------------------------------------------------------------------
__global__ __launch_bounds__(256) void numerator_kernel(
    const float* __restrict__ em, const float* __restrict__ E1,
    const float* __restrict__ E2, const int* __restrict__ targets,
    const unsigned char* __restrict__ masks, float* __restrict__ numer)
{
    __shared__ float red[256];
    const int b = blockIdx.x, tid = threadIdx.x;
    float acc = 0.f;
    for (int t = tid; t < TT; t += 256) {
        const int tg = targets[b * TT + t];
        float s = em[((size_t)(b * TT + t)) * VV + tg];
        if (t > 0) {
            const int tp = targets[b * TT + t - 1];
            const float* __restrict__ e1 = E1 + (size_t)tp * RNK;
            const float* __restrict__ e2 = E2 + (size_t)tg * RNK;
            float d = 0.f;
#pragma unroll
            for (int r = 0; r < RNK; ++r) d = fmaf(e1[r], e2[r], d);
            s += d;
        }
        if (masks[b * TT + t]) acc += s;
    }
    red[tid] = acc;
    __syncthreads();
    for (int off = 128; off > 0; off >>= 1) {
        if (tid < off) red[tid] += red[tid + off];
        __syncthreads();
    }
    if (tid == 0) numer[b] = red[0];
}

// ---------------------------------------------------------------------------
// Kernel 3: fused beam-transition GEMM (v_wmma_f32_16x16x32_bf16, K==RANK==32)
// + forward logsumexp scan. One workgroup (8 waves) per batch element.
// E1/E2 beam rows staged in LDS as bf16 (converted once), fragments read as
// packed 32-bit pairs; column lse done by 4 threads per beam column.
// ---------------------------------------------------------------------------
__global__ __launch_bounds__(256) void crf_scan_kernel(
    const float* __restrict__ E1, const float* __restrict__ E2,
    const int* __restrict__ beam_tgts, const float* __restrict__ beam_em,
    const unsigned char* __restrict__ masks, const float* __restrict__ numer,
    float* __restrict__ out)
{
    __shared__ unsigned short Abf[BM * RNK];   // bf16 E1 rows of prev beam (4KB)
    __shared__ unsigned short Bbf[BM * RNK];   // bf16 E2 rows of next beam (4KB)
    __shared__ float trans[BM * BM];           // 64x64 transition logits (16KB)
    __shared__ float score[BM];
    __shared__ float red[256];
    __shared__ float colmax[BM];
    __shared__ int   prevb[BM], nextb[BM];
    __shared__ float emitv[BM];

    const int b    = blockIdx.x;
    const int tid  = threadIdx.x;
    const int wave = tid >> 5, lane = tid & 31;
    const int mm = lane & 15, hi = lane >> 4;
    const int l  = tid & 63;                   // beam column owned in reduction
    const int k0 = (tid >> 6) * 16;            // k-range start for this thread

    if (tid < BM) score[tid] = beam_em[(b * TT + 0) * BM + tid];
    __syncthreads();

    for (int t = 1; t < TT; ++t) {
        if (tid < BM) {
            prevb[tid] = beam_tgts[(b * TT + t - 1) * BM + tid];
            nextb[tid] = beam_tgts[(b * TT + t    ) * BM + tid];
            emitv[tid] = beam_em  [(b * TT + t    ) * BM + tid];
        }
        __syncthreads();

        // gather + one-time f32->bf16 convert (E1/E2 are L2-resident, 4MB each)
        for (int i = tid; i < BM * RNK; i += 256) {
            Abf[i] = f2bf(E1[(size_t)prevb[i >> 5] * RNK + (i & 31)]);
            Bbf[i] = f2bf(E2[(size_t)nextb[i >> 5] * RNK + (i & 31)]);
        }
        __syncthreads();

        // 16 WMMA tiles (4x4 grid of 16x16), 2 per wave; K = 32 in one WMMA
#pragma unroll
        for (int rep = 0; rep < 2; ++rep) {
            const int tile = wave * 2 + rep;
            const int ti = tile >> 2, tj = tile & 3;
            union { unsigned u[8]; v16bf v; } ua, ub;
#pragma unroll
            for (int p = 0; p < 8; ++p) {
                const int e = 2 * p;
                // A (16x32 MxK): lane<16 holds K {0..7,16..23}; lane>=16 K {8..15,24..31}
                const int kkA = hi * 8 + (e < 8 ? e : e + 8);
                ua.u[p] = *(const unsigned*)&Abf[(ti * 16 + mm) * RNK + kkA];
                // B (32x16 KxN): lane<16 holds K {0..7,16..23}; lane>=16 K {8..15,24..31}
                const int kkB = hi * 8 + (e < 8 ? e : e + 8);
                ub.u[p] = *(const unsigned*)&Bbf[(tj * 16 + mm) * RNK + kkB];
            }
            v8f c = {};
            c = __builtin_amdgcn_wmma_f32_16x16x32_bf16(
                    false, ua.v, false, ub.v, (short)0, c, false, false);
#pragma unroll
            for (int j = 0; j < 8; ++j) {
                const int m = j + hi * 8;        // D: VGPR j -> row j (lanes 0-15) / j+8
                trans[(ti * 16 + m) * BM + tj * 16 + mm] = c[j];
            }
        }
        __syncthreads();

        // ns[l] = lse_k(score[k] + trans[k,l]) + emit[l]; 4 threads per column
        float pm = -INFINITY;
#pragma unroll
        for (int kk = 0; kk < 16; ++kk)
            pm = fmaxf(pm, score[k0 + kk] + trans[(k0 + kk) * BM + l]);
        red[tid] = pm;
        __syncthreads();
        if (tid < BM)
            colmax[tid] = fmaxf(fmaxf(red[tid], red[tid + 64]),
                                fmaxf(red[tid + 128], red[tid + 192]));
        __syncthreads();
        const float cm = colmax[l];
        float ps = 0.f;
#pragma unroll
        for (int kk = 0; kk < 16; ++kk)
            ps += __expf(score[k0 + kk] + trans[(k0 + kk) * BM + l] - cm);
        red[tid] = ps;
        __syncthreads();
        float ns = 0.f;
        if (tid < BM) {
            float s = red[tid] + red[tid + 64] + red[tid + 128] + red[tid + 192];
            ns = colmax[tid] + __logf(s) + emitv[tid];
        }
        __syncthreads();
        if (tid < BM && masks[b * TT + t]) score[tid] = ns;
        __syncthreads();
    }

    if (tid == 0) {
        float mx = -INFINITY;
        for (int k = 0; k < BM; ++k) mx = fmaxf(mx, score[k]);
        float s = 0.f;
        for (int k = 0; k < BM; ++k) s += __expf(score[k] - mx);
        out[b] = numer[b] - (mx + __logf(s));
    }
}

// ---------------------------------------------------------------------------
extern "C" void kernel_launch(void* const* d_in, const int* in_sizes, int n_in,
                              void* d_out, int out_size, void* d_ws, size_t ws_size,
                              hipStream_t stream) {
    const float*         emissions = (const float*)d_in[0];
    const float*         E1        = (const float*)d_in[1];
    const float*         E2        = (const float*)d_in[2];
    const int*           targets   = (const int*)d_in[3];
    const unsigned char* masks     = (const unsigned char*)d_in[4];
    float*               out       = (float*)d_out;

    // workspace layout: beam_tgts (1MB) | beam_em (1MB) | numer (64B)
    char* ws = (char*)d_ws;
    int*   beam_tgts = (int*)ws;
    float* beam_em   = (float*)(ws + (size_t)BB * TT * BM * sizeof(int));
    float* numer     = (float*)(ws + 2 * (size_t)BB * TT * BM * sizeof(int));

    topk64_kernel   <<<BB * TT, 256, 0, stream>>>(emissions, targets, beam_tgts, beam_em);
    numerator_kernel<<<BB,      256, 0, stream>>>(emissions, E1, E2, targets, masks, numer);
    crf_scan_kernel <<<BB,      256, 0, stream>>>(E1, E2, beam_tgts, beam_em, masks, numer, out);
}
